// nODE_7284264534323
// MI455X (gfx1250) — compile-verified, matrix operand order checked by reading, not run on hardware
//
#include <hip/hip_runtime.h>
#include <hip/hip_bf16.h>
#include <math.h>

typedef __bf16 bf16_t;
typedef __attribute__((ext_vector_type(16))) __bf16 v16bf;
typedef __attribute__((ext_vector_type(8)))  __bf16 v8bf;
typedef __attribute__((ext_vector_type(8)))  float  v8f;

#define ODE_DIM   512
#define BATCHSZ   32768
#define NSTEPS    50
#define DT_F      (1.0f / 50.0f)

#define MWG       64            // rows resident per workgroup
#define NTHREADS  512           // 16 waves (wave32)
#define SF_STRIDE 516           // f32 state row stride (pad: bank + 16B align)
#define SB_STRIDE 520           // bf16 state row stride (pad: 16B align)

// hardware tanh (gfx1250 v_tanh_f32) with libm fallback
static __device__ inline float fast_tanh(float v) {
#if defined(__has_builtin)
#if __has_builtin(__builtin_amdgcn_tanhf)
    return __builtin_amdgcn_tanhf(v);
#else
    return tanhf(v);
#endif
#else
    return tanhf(v);
#endif
}

// --- W f32 -> bf16 conversion (runs once per launch) ---
__global__ void w_to_bf16_kernel(const float* __restrict__ W,
                                 bf16_t* __restrict__ Wb, int n) {
    int i = blockIdx.x * blockDim.x + threadIdx.x;
    if (i < n) Wb[i] = (bf16_t)W[i];
}

static __device__ inline v16bf cat8(v8bf lo, v8bf hi) {
    return __builtin_shufflevector(lo, hi, 0,1,2,3,4,5,6,7,8,9,10,11,12,13,14,15);
}

__global__ __launch_bounds__(NTHREADS, 1)
void node_euler_kernel(const float* __restrict__ x,
                       const bf16_t* __restrict__ Wb,
                       const float* __restrict__ b_in,
                       const float* __restrict__ gamma,
                       float* __restrict__ out) {
    __shared__ float  sf[MWG * SF_STRIDE];   // f32 state, 64 rows
    __shared__ bf16_t sb[MWG * SB_STRIDE];   // bf16 shadow for WMMA A

    const int tid  = threadIdx.x;
    const int lane = tid & 31;
    const int wave = tid >> 5;               // 0..15
    const int lcol = lane & 15;              // column / row-in-tile index
    const int g    = lane >> 4;              // half-wave selector
    const int row0 = blockIdx.x * MWG;       // global batch row base

    // cooperative global->LDS load of x (8 threads per row, 64 f32 each)
    const int lrow = tid >> 3;
    const int c0   = (tid & 7) * 64;
    {
        const float4* gx = (const float4*)(x + (size_t)(row0 + lrow) * ODE_DIM + c0);
#pragma unroll
        for (int j = 0; j < 16; ++j)
            *(float4*)&sf[lrow * SF_STRIDE + c0 + j * 4] = gx[j];
    }

    // per-lane bias / gamma for this wave's two N-tiles
    float bv[2], gv[2];
#pragma unroll
    for (int j = 0; j < 2; ++j) {
        int ncol = (2 * wave + j) * 16 + lcol;
        bv[j] = b_in[ncol];
        gv[j] = gamma[ncol];
    }
    __syncthreads();

    for (int step = 0; step < NSTEPS; ++step) {
        // ---- phase 1: refresh bf16 shadow of the state ----
#pragma unroll
        for (int j = 0; j < 8; ++j) {
            v8bf o;
#pragma unroll
            for (int i = 0; i < 8; ++i)
                o[i] = (bf16_t)sf[lrow * SF_STRIDE + c0 + j * 8 + i];
            *(v8bf*)&sb[lrow * SB_STRIDE + c0 + j * 8] = o;
        }
        __syncthreads();

        // ---- phase 2: Y = S @ W^T via WMMA bf16, f32 accumulate ----
        v8f acc[4][2];
#pragma unroll
        for (int mt = 0; mt < 4; ++mt)
#pragma unroll
            for (int j = 0; j < 2; ++j)
                acc[mt][j] = (v8f){0.f,0.f,0.f,0.f,0.f,0.f,0.f,0.f};

#pragma unroll 2
        for (int k0 = 0; k0 < ODE_DIM; k0 += 32) {
            // A fragments: 4 M-tiles from LDS (lane&15 = row, K half-chunks by g)
            v16bf afr[4];
#pragma unroll
            for (int mt = 0; mt < 4; ++mt) {
                const bf16_t* ap = &sb[(mt * 16 + lcol) * SB_STRIDE + k0 + g * 8];
                afr[mt] = cat8(*(const v8bf*)ap, *(const v8bf*)(ap + 16));
            }
            // B fragments: 2 N-tiles streamed from W (row n of W = column n of W^T)
#pragma unroll
            for (int j = 0; j < 2; ++j) {
                const bf16_t* bp = Wb + (size_t)((2 * wave + j) * 16 + lcol) * ODE_DIM
                                   + k0 + g * 8;
                v16bf bfr = cat8(*(const v8bf*)bp, *(const v8bf*)(bp + 16));
#pragma unroll
                for (int mt = 0; mt < 4; ++mt)
                    acc[mt][j] = __builtin_amdgcn_wmma_f32_16x16x32_bf16(
                        false, afr[mt], false, bfr, (short)0, acc[mt][j],
                        false, false);
            }
        }

        // ---- epilogue: s += DT*(s*gamma + tanh(acc + b)); columns are
        //      partitioned per wave, so no cross-wave write hazard ----
#pragma unroll
        for (int mt = 0; mt < 4; ++mt) {
#pragma unroll
            for (int j = 0; j < 2; ++j) {
                int ncol = (2 * wave + j) * 16 + lcol;
#pragma unroll
                for (int r = 0; r < 8; ++r) {
                    int mrow = mt * 16 + r + g * 8;
                    float* sp = &sf[mrow * SF_STRIDE + ncol];
                    float s  = *sp;
                    float th = fast_tanh(acc[mt][j][r] + bv[j]);
                    *sp = s + DT_F * (s * gv[j] + th);
                }
            }
        }
        __syncthreads();
    }

    // cooperative LDS->global store of the final state
    {
        float4* gy = (float4*)(out + (size_t)(row0 + lrow) * ODE_DIM + c0);
#pragma unroll
        for (int j = 0; j < 16; ++j)
            gy[j] = *(const float4*)&sf[lrow * SF_STRIDE + c0 + j * 4];
    }
}

extern "C" void kernel_launch(void* const* d_in, const int* in_sizes, int n_in,
                              void* d_out, int out_size, void* d_ws, size_t ws_size,
                              hipStream_t stream) {
    const float* x     = (const float*)d_in[0];
    const float* W_in  = (const float*)d_in[1];
    const float* b_in  = (const float*)d_in[2];
    const float* gamma = (const float*)d_in[3];
    float*       out   = (float*)d_out;
    bf16_t*      Wb    = (bf16_t*)d_ws;     // 512*512*2 = 512 KB scratch

    w_to_bf16_kernel<<<(ODE_DIM * ODE_DIM) / 256, 256, 0, stream>>>(
        W_in, Wb, ODE_DIM * ODE_DIM);

    node_euler_kernel<<<BATCHSZ / MWG, NTHREADS, 0, stream>>>(
        x, Wb, b_in, gamma, out);
}
